// CrossAttentionBlock_56642028700071
// MI455X (gfx1250) — matrix operand order repaired and checked by reading, not compiled, not run on hardware
//
#include <hip/hip_runtime.h>
#include <hip/hip_bf16.h>

// ---------------- types ----------------
typedef __bf16 bf16_t;
typedef __bf16  v16bf __attribute__((ext_vector_type(16)));
typedef float   v8f   __attribute__((ext_vector_type(8)));

#define B_   8
#define C_   512
#define N_   1024      // 32*32 tokens
#define H_   8
#define DK_  64
#define INNER_ 512     // H_*DK_
// q pre-scale: dk^-0.5 * log2(e)  -> softmax done in exp2 domain
#define QSCALE (0.125f * 1.44269504088896340736f)

// ---------------- WMMA fragment helpers (CDNA5 ISA 7.12.2) ----------------
// A 16x32 bf16, row-major source, lda in elements (must be mult of 8).
// lane 0-15: M=lane, K = 0..15; lane 16-31: M=lane-16, K = 16..31
__device__ __forceinline__ v16bf load_a_frag(const bf16_t* A, int lda) {
    const int lane = threadIdx.x & 31;
    const int g = lane >> 4, m = lane & 15;
    const bf16_t* p = A + m * lda + 16 * g;
    union { uint4 u[2]; v16bf v; } u;
    u.u[0] = *reinterpret_cast<const uint4*>(p);        // K = 16g .. 16g+7
    u.u[1] = *reinterpret_cast<const uint4*>(p + 8);    // K = 16g+8 .. 16g+15
    return u.v;
}

// B 32x16 bf16 from a [n][k] row-major matrix (i.e. weight [out,in]), ldb elems.
// lane 0-15: N=lane, K=0..15 ; lane 16-31: N=lane-16, K=16..31 (contiguous along k)
__device__ __forceinline__ v16bf load_b_frag(const bf16_t* Bp, int ldb) {
    const int lane = threadIdx.x & 31;
    const int g = lane >> 4, n = lane & 15;
    const bf16_t* p = Bp + n * ldb + 16 * g;
    union { uint4 u[2]; v16bf v; } u;
    u.u[0] = *reinterpret_cast<const uint4*>(p);
    u.u[1] = *reinterpret_cast<const uint4*>(p + 8);
    return u.v;
}

__device__ __forceinline__ v8f wmma_bf16(v16bf a, v16bf b, v8f c) {
    return __builtin_amdgcn_wmma_f32_16x16x32_bf16(false, a, false, b,
                                                   (short)0, c, false, false);
}

__device__ __forceinline__ void wait_lds() {
#if __has_builtin(__builtin_amdgcn_s_wait_dscnt)
    __builtin_amdgcn_s_wait_dscnt(0);
#else
    asm volatile("s_wait_dscnt 0" ::: "memory");
#endif
}

// ---------------- stage 0: conversions ----------------
__global__ __launch_bounds__(256) void k_cvt_bf16(const float* __restrict__ s,
                                                  bf16_t* __restrict__ d, int n) {
    int i = blockIdx.x * 256 + threadIdx.x;
    if (i < n) d[i] = (bf16_t)s[i];
}

// x [B,C,N] (NCHW with HW flattened) -> xt [B,N,C] bf16
__global__ __launch_bounds__(256) void k_make_xt(const float* __restrict__ x,
                                                 bf16_t* __restrict__ xt) {
    int i = blockIdx.x * 256 + threadIdx.x;           // B*N*C threads, c fastest
    int c = i & (C_ - 1);
    int t = i >> 9;                                   // /C_
    int n = t & (N_ - 1);
    int b = t >> 10;                                  // /N_
    xt[i] = (bf16_t)x[(b * C_ + c) * N_ + n];
}

// ---------------- stage 1: fused Q/K/V projection ----------------
// one wave computes a 16x64 output strip (A-fragment reused across 4 B tiles).
// strips 0..7 -> Q (pre-scaled), 8..15 -> K, 16..23 -> V (stored [B,H,DK,N])
__global__ __launch_bounds__(256) void k_qkv_proj(
    const bf16_t* __restrict__ xt, const bf16_t* __restrict__ Wq,
    const float* __restrict__ bq, const bf16_t* __restrict__ Wkv,
    const float* __restrict__ bkv, bf16_t* __restrict__ qb,
    bf16_t* __restrict__ kb, bf16_t* __restrict__ vT) {
    const int wave = blockIdx.x * 8 + (threadIdx.x >> 5);
    const int st = wave % 24;            // 64-wide output strip (1536/64)
    const int t  = wave / 24;
    const int nt = t & 63;               // row tile (N/16)
    const int b  = t >> 6;
    if (b >= B_) return;                 // wave-uniform

    const int o0 = st * 64;
    const bf16_t* A0 = xt + (b * N_ + nt * 16) * C_;
    const bf16_t* W;
    const float* bias;
    if (o0 < 512) { W = Wq + o0 * C_;           bias = bq + o0; }
    else          { W = Wkv + (o0 - 512) * C_;  bias = bkv + (o0 - 512); }

    v8f a0 = {}, a1 = {}, a2 = {}, a3 = {};
#pragma unroll
    for (int k0 = 0; k0 < C_; k0 += 32) {
        const v16bf af = load_a_frag(A0 + k0, C_);
        a0 = wmma_bf16(af, load_b_frag(W + 0 * 16 * C_ + k0, C_), a0);
        a1 = wmma_bf16(af, load_b_frag(W + 1 * 16 * C_ + k0, C_), a1);
        a2 = wmma_bf16(af, load_b_frag(W + 2 * 16 * C_ + k0, C_), a2);
        a3 = wmma_bf16(af, load_b_frag(W + 3 * 16 * C_ + k0, C_), a3);
    }
    v8f acc[4] = {a0, a1, a2, a3};

    const int lane = threadIdx.x & 31;
    const int g = lane >> 4, nn = lane & 15;

    if (o0 < 512) {                       // Q, pre-scaled by 1/sqrt(dk)*log2e
        const int h = o0 >> 6;
        bf16_t* dst = qb + ((b * H_ + h) * N_ + nt * 16) * DK_;
#pragma unroll
        for (int tt = 0; tt < 4; ++tt) {
            const float bv = bias[16 * tt + nn];
#pragma unroll
            for (int r = 0; r < 8; ++r)
                dst[(r + 8 * g) * DK_ + 16 * tt + nn] =
                    (bf16_t)((acc[tt][r] + bv) * QSCALE);
        }
    } else if (o0 < 1024) {               // K
        const int h = (o0 - 512) >> 6;
        bf16_t* dst = kb + ((b * H_ + h) * N_ + nt * 16) * DK_;
#pragma unroll
        for (int tt = 0; tt < 4; ++tt) {
            const float bv = bias[16 * tt + nn];
#pragma unroll
            for (int r = 0; r < 8; ++r)
                dst[(r + 8 * g) * DK_ + 16 * tt + nn] = (bf16_t)(acc[tt][r] + bv);
        }
    } else {                              // V -> transposed [b][h][d][n]
        const int h = (o0 - 1024) >> 6;
        bf16_t* dst = vT + (b * H_ + h) * DK_ * N_ + nt * 16;
#pragma unroll
        for (int tt = 0; tt < 4; ++tt) {
            const float bv = bias[16 * tt + nn];
#pragma unroll
            for (int r = 0; r < 8; ++r)
                dst[(16 * tt + nn) * N_ + r + 8 * g] = (bf16_t)(acc[tt][r] + bv);
        }
    }
}

// ---------------- stage 2: flash attention ----------------
// one wave = one (b,h,16-row query block); stream keys in chunks of 64
__global__ __launch_bounds__(256) void k_attn(
    const bf16_t* __restrict__ qb, const bf16_t* __restrict__ kb,
    const bf16_t* __restrict__ vT, bf16_t* __restrict__ res) {
    __shared__ __align__(16) bf16_t pstage[8][16 * 64];   // per-wave P staging
    const int wv = threadIdx.x >> 5;
    const int wave = blockIdx.x * 8 + wv;
    const int it = wave & 63;            // query row tile
    const int bh = wave >> 6;            // b*H + h
    const bf16_t* Q = qb + (bh * N_ + it * 16) * DK_;
    const bf16_t* K = kb + bh * N_ * DK_;
    const bf16_t* V = vT + bh * DK_ * N_;     // [DK][N]

    const v16bf qa0 = load_a_frag(Q, DK_);        // d = 0..31
    const v16bf qa1 = load_a_frag(Q + 32, DK_);   // d = 32..63

    const int lane = threadIdx.x & 31;
    const int g = lane >> 4, nn = lane & 15;
    float mrun[8], lrun[8];
#pragma unroll
    for (int r = 0; r < 8; ++r) { mrun[r] = -3.0e38f; lrun[r] = 0.f; }
    v8f o0 = {}, o1 = {}, o2 = {}, o3 = {};
    bf16_t* pbuf = pstage[wv];

    for (int j0 = 0; j0 < N_; j0 += 64) {
        if (j0 + 64 < N_) {               // pull next K/V chunk toward the WGP
            __builtin_prefetch(K + (j0 + 64) * DK_ + lane * 2 * DK_, 0, 3);
            __builtin_prefetch(V + lane * 2 * N_ + (j0 + 64), 0, 3);
        }
        v8f s[4] = {{}, {}, {}, {}};
#pragma unroll
        for (int tt = 0; tt < 4; ++tt) {
            s[tt] = wmma_bf16(qa0, load_b_frag(K + (j0 + 16 * tt) * DK_, DK_), s[tt]);
            s[tt] = wmma_bf16(qa1, load_b_frag(K + (j0 + 16 * tt) * DK_ + 32, DK_), s[tt]);
        }

        // online softmax (exp2 domain; rows live across 16 lanes of a half-wave)
#pragma unroll
        for (int r = 0; r < 8; ++r) {
            const float v0 = s[0][r], v1 = s[1][r], v2 = s[2][r], v3 = s[3][r];
            float mx = fmaxf(fmaxf(v0, v1), fmaxf(v2, v3));
            mx = fmaxf(mx, __shfl_xor(mx, 1));
            mx = fmaxf(mx, __shfl_xor(mx, 2));
            mx = fmaxf(mx, __shfl_xor(mx, 4));
            mx = fmaxf(mx, __shfl_xor(mx, 8));
            const float mn = fmaxf(mrun[r], mx);
            const float corr = __builtin_amdgcn_exp2f(mrun[r] - mn);
            const float p0 = __builtin_amdgcn_exp2f(v0 - mn);
            const float p1 = __builtin_amdgcn_exp2f(v1 - mn);
            const float p2 = __builtin_amdgcn_exp2f(v2 - mn);
            const float p3 = __builtin_amdgcn_exp2f(v3 - mn);
            float rs = (p0 + p1) + (p2 + p3);
            rs += __shfl_xor(rs, 1);
            rs += __shfl_xor(rs, 2);
            rs += __shfl_xor(rs, 4);
            rs += __shfl_xor(rs, 8);
            lrun[r] = lrun[r] * corr + rs;
            mrun[r] = mn;
            o0[r] *= corr; o1[r] *= corr; o2[r] *= corr; o3[r] *= corr;
            const int row = r + 8 * g;
            pbuf[row * 64 + nn]      = (bf16_t)p0;
            pbuf[row * 64 + 16 + nn] = (bf16_t)p1;
            pbuf[row * 64 + 32 + nn] = (bf16_t)p2;
            pbuf[row * 64 + 48 + nn] = (bf16_t)p3;
        }
        wait_lds();                                  // same-wave LDS RAW
        const v16bf pa0 = load_a_frag(pbuf, 64);     // keys j0    .. j0+31
        const v16bf pa1 = load_a_frag(pbuf + 32, 64);// keys j0+32 .. j0+63
        o0 = wmma_bf16(pa0, load_b_frag(V + 0 * 16 * N_ + j0, N_), o0);
        o0 = wmma_bf16(pa1, load_b_frag(V + 0 * 16 * N_ + j0 + 32, N_), o0);
        o1 = wmma_bf16(pa0, load_b_frag(V + 1 * 16 * N_ + j0, N_), o1);
        o1 = wmma_bf16(pa1, load_b_frag(V + 1 * 16 * N_ + j0 + 32, N_), o1);
        o2 = wmma_bf16(pa0, load_b_frag(V + 2 * 16 * N_ + j0, N_), o2);
        o2 = wmma_bf16(pa1, load_b_frag(V + 2 * 16 * N_ + j0 + 32, N_), o2);
        o3 = wmma_bf16(pa0, load_b_frag(V + 3 * 16 * N_ + j0, N_), o3);
        o3 = wmma_bf16(pa1, load_b_frag(V + 3 * 16 * N_ + j0 + 32, N_), o3);
    }

    // finalize: res[b][n][h*64+d] bf16
    const int b = bh >> 3, h = bh & 7;
    float inv[8];
#pragma unroll
    for (int r = 0; r < 8; ++r) inv[r] = __builtin_amdgcn_rcpf(lrun[r]);
#pragma unroll
    for (int r = 0; r < 8; ++r) {
        const int row = it * 16 + r + 8 * g;
        bf16_t* dst = res + (b * N_ + row) * INNER_ + h * DK_ + nn;
        dst[0]  = (bf16_t)(o0[r] * inv[r]);
        dst[16] = (bf16_t)(o1[r] * inv[r]);
        dst[32] = (bf16_t)(o2[r] * inv[r]);
        dst[48] = (bf16_t)(o3[r] * inv[r]);
    }
}

// ---------------- stage 3: output projection (16x64 strips), f32 out ----------------
__global__ __launch_bounds__(256) void k_out_proj(
    const bf16_t* __restrict__ res, const bf16_t* __restrict__ Wp,
    const float* __restrict__ bp, float* __restrict__ out) {
    const int wave = blockIdx.x * 8 + (threadIdx.x >> 5);
    const int st = wave & 7;             // 64-wide col strip (C/64)
    const int mt = wave >> 3;            // (B*N)/16 row tiles
    const bf16_t* A0 = res + mt * 16 * INNER_;
    const bf16_t* W = Wp + st * 64 * INNER_;
    v8f a0 = {}, a1 = {}, a2 = {}, a3 = {};
#pragma unroll
    for (int k0 = 0; k0 < INNER_; k0 += 32) {
        const v16bf af = load_a_frag(A0 + k0, INNER_);
        a0 = wmma_bf16(af, load_b_frag(W + 0 * 16 * INNER_ + k0, INNER_), a0);
        a1 = wmma_bf16(af, load_b_frag(W + 1 * 16 * INNER_ + k0, INNER_), a1);
        a2 = wmma_bf16(af, load_b_frag(W + 2 * 16 * INNER_ + k0, INNER_), a2);
        a3 = wmma_bf16(af, load_b_frag(W + 3 * 16 * INNER_ + k0, INNER_), a3);
    }
    v8f acc[4] = {a0, a1, a2, a3};
    const int lane = threadIdx.x & 31;
    const int g = lane >> 4, nn = lane & 15;
#pragma unroll
    for (int tt = 0; tt < 4; ++tt) {
        const float bv = bp[st * 64 + 16 * tt + nn];
#pragma unroll
        for (int r = 0; r < 8; ++r)
            out[(mt * 16 + r + 8 * g) * C_ + st * 64 + 16 * tt + nn] =
                acc[tt][r] + bv;
    }
}

// ---------------- launch ----------------
extern "C" void kernel_launch(void* const* d_in, const int* in_sizes, int n_in,
                              void* d_out, int out_size, void* d_ws, size_t ws_size,
                              hipStream_t stream) {
    (void)in_sizes; (void)n_in; (void)out_size; (void)ws_size;
    const float* x   = (const float*)d_in[0];
    // d_in[1] = y : unused by reference
    const float* Wq  = (const float*)d_in[2];
    const float* bq  = (const float*)d_in[3];
    const float* Wkv = (const float*)d_in[4];
    const float* bkv = (const float*)d_in[5];
    const float* Wp  = (const float*)d_in[6];
    const float* bp  = (const float*)d_in[7];
    float* out = (float*)d_out;

    char* ws = (char*)d_ws;
    size_t off = 0;
    auto alloc = [&](size_t bytes) -> void* {
        void* p = (void*)(ws + off);
        off += (bytes + 255) & ~(size_t)255;
        return p;
    };
    bf16_t* xt    = (bf16_t*)alloc((size_t)B_ * N_ * C_ * 2);       // 8 MB
    bf16_t* Wq_b  = (bf16_t*)alloc((size_t)INNER_ * C_ * 2);
    bf16_t* Wkv_b = (bf16_t*)alloc((size_t)2 * INNER_ * C_ * 2);
    bf16_t* Wp_b  = (bf16_t*)alloc((size_t)C_ * INNER_ * 2);
    bf16_t* qb    = (bf16_t*)alloc((size_t)B_ * H_ * N_ * DK_ * 2); // 8 MB
    bf16_t* kb    = (bf16_t*)alloc((size_t)B_ * H_ * N_ * DK_ * 2);
    bf16_t* vT    = (bf16_t*)alloc((size_t)B_ * H_ * DK_ * N_ * 2);
    bf16_t* resb  = (bf16_t*)alloc((size_t)B_ * N_ * INNER_ * 2);

    k_cvt_bf16<<<(INNER_ * C_) / 256, 256, 0, stream>>>(Wq, Wq_b, INNER_ * C_);
    k_cvt_bf16<<<(2 * INNER_ * C_) / 256, 256, 0, stream>>>(Wkv, Wkv_b, 2 * INNER_ * C_);
    k_cvt_bf16<<<(C_ * INNER_) / 256, 256, 0, stream>>>(Wp, Wp_b, C_ * INNER_);
    k_make_xt<<<(B_ * N_ * C_) / 256, 256, 0, stream>>>(x, xt);

    // 8 * 64 * 24 = 12288 strip-waves, 8 waves/block
    k_qkv_proj<<<1536, 256, 0, stream>>>(xt, Wq_b, bq, Wkv_b, bkv, qb, kb, vT);
    // 8*8*64 = 4096 waves
    k_attn<<<512, 256, 0, stream>>>(qb, kb, vT, resb);
    // 512 row tiles * 8 strips = 4096 waves
    k_out_proj<<<512, 256, 0, stream>>>(resb, Wp_b, bp, out);
}
